// AdapAFPL1_80376017977883
// MI455X (gfx1250) — matrix operand order repaired and checked by reading, not compiled, not run on hardware
//
#include <hip/hip_runtime.h>
#include <hip/hip_bf16.h>

#define N_NODES 50000
#define N_EDGES 800000
#define NB      256
#define DIN     64
#define DH      128
#define DED     16
#define DOUT    8

typedef __attribute__((ext_vector_type(16))) _Float16 v16h;
typedef __attribute__((ext_vector_type(8)))  float    v8f;

__device__ __forceinline__ float lrelu(float v){ return v > 0.f ? v : 0.01f * v; }
__device__ __forceinline__ float sigm(float v){ return 1.f / (1.f + __expf(-v)); }
__device__ __forceinline__ float eluf(float v){ return v > 0.f ? v : (__expf(v) - 1.f); }

// ---------------- WMMA fragment loaders (contiguous k = 16*half + e, wave32) ----------------
// A 16x32 f16, row-major src, lda in elements.  lane<16: row=lane, K = k0+0..15
// lane>=16: row=lane-16, K = k0+16..31
__device__ __forceinline__ v16h load_a_f16(const _Float16* A, int lda, int row0, int k0, int M){
  int lane = threadIdx.x & 31;
  int half = lane >> 4;
  int m    = row0 + (lane & 15);
  v16h f;
  if (m < M) {
    const _Float16* p = A + (size_t)m * lda + k0 + 16 * half;
#pragma unroll
    for (int e = 0; e < 8; ++e)  f[e] = p[e];
#pragma unroll
    for (int e = 8; e < 16; ++e) f[e] = p[e];
  } else {
#pragma unroll
    for (int e = 0; e < 16; ++e) f[e] = (_Float16)0.f;
  }
  return f;
}

// Same layout, but source is f32 with fused ELU (for GRU input x = elu(h)).
__device__ __forceinline__ v16h load_a_elu_f32(const float* A, int lda, int row0, int k0, int M){
  int lane = threadIdx.x & 31;
  int half = lane >> 4;
  int m    = row0 + (lane & 15);
  v16h f;
  if (m < M) {
    const float* p = A + (size_t)m * lda + k0 + 16 * half;
#pragma unroll
    for (int e = 0; e < 8; ++e)  f[e] = (_Float16)eluf(p[e]);
#pragma unroll
    for (int e = 8; e < 16; ++e) f[e] = (_Float16)eluf(p[e]);
  } else {
#pragma unroll
    for (int e = 0; e < 16; ++e) f[e] = (_Float16)0.f;
  }
  return f;
}

// B 32x16 f16 built from row-major weight W[Nout,K] (B = W^T).
// lane<16: col=n0+lane, K = k0+0..15 ; lane>=16: col=n0+lane-16, K = k0+16..31
__device__ __forceinline__ v16h load_b_f16(const _Float16* W, int ldw, int n0, int k0){
  int lane  = threadIdx.x & 31;
  int khalf = lane >> 4;
  int n     = n0 + (lane & 15);
  const _Float16* p = W + (size_t)n * ldw + k0 + 16 * khalf;
  v16h f;
#pragma unroll
  for (int e = 0; e < 16; ++e) f[e] = p[e];
  return f;
}

__device__ __forceinline__ v8f wmma32(v16h a, v16h b, v8f c){
  return __builtin_amdgcn_wmma_f32_16x16x32_f16(false, a, false, b, (short)0, c, false, false);
}

// ---------------- small utility kernels ----------------
__global__ void cast_pad_kernel(const float* __restrict__ src, _Float16* __restrict__ dst,
                                int rows, int scols, int dcols){
  int i = blockIdx.x * blockDim.x + threadIdx.x;
  if (i >= rows * dcols) return;
  int r = i / dcols, c = i - r * dcols;
  dst[i] = (c < scols) ? (_Float16)src[(size_t)r * scols + c] : (_Float16)0.f;
}

__global__ void fill_kernel(float* p, float v, int n){
  int i = blockIdx.x * blockDim.x + threadIdx.x;
  if (i < n) p[i] = v;
}

__global__ void bias_init_kernel(float* __restrict__ p, const float* __restrict__ bias,
                                 int rows, int cols){
  int i = blockIdx.x * blockDim.x + threadIdx.x;
  if (i >= rows * cols) return;
  p[i] = bias[i % cols];
}

// nw = sigmoid(x @ lin_node_w + b); xin[:, :64] = f16(x * nw)
__global__ void node_gate_kernel(const float* __restrict__ x, const float* __restrict__ w,
                                 const float* __restrict__ b, _Float16* __restrict__ xin, int N){
  int n = blockIdx.x * blockDim.x + threadIdx.x;
  if (n >= N) return;
  const float* xr = x + (size_t)n * DIN;
  float acc = b[0];
#pragma unroll
  for (int k = 0; k < DIN; ++k) acc += xr[k] * w[k];
  float g = sigm(acc);
  _Float16* d = xin + (size_t)n * 192;
#pragma unroll
  for (int k = 0; k < DIN; ++k) d[k] = (_Float16)(xr[k] * g);
}

// ---------------- generic WMMA GEMM: C = act(A_f16[M,K] @ W_f16[Nout,K]^T + bias) ----------------
// act: 0=none, 1=leaky_relu.  Cf (f32) and Ch (f16) outputs optional.
__global__ __launch_bounds__(256)
void gemm_f16_kernel(const _Float16* __restrict__ A, int lda,
                     const _Float16* __restrict__ W, int ldw,
                     const float* __restrict__ bias,
                     float* __restrict__ Cf, int ldc,
                     _Float16* __restrict__ Ch, int ldch,
                     int M, int Nout, int K, int act){
  int wave = threadIdx.x >> 5, lane = threadIdx.x & 31;
  int row0 = blockIdx.x * 16;
  int n0   = blockIdx.y * 128 + wave * 16;
  if (n0 >= Nout) return;
  v8f acc = {};
  for (int k0 = 0; k0 < K; k0 += 32){
    v16h a = load_a_f16(A, lda, row0, k0, M);
    v16h b = load_b_f16(W, ldw, n0, k0);
    acc = wmma32(a, b, acc);
  }
  int half = lane >> 4, n = n0 + (lane & 15);
  float bv = bias ? bias[n] : 0.f;
#pragma unroll
  for (int r = 0; r < 8; ++r){
    int m = row0 + r + 8 * half;
    if (m < M){
      float v = acc[r] + bv;
      if (act == 1) v = lrelu(v);
      if (Cf) Cf[(size_t)m * ldc  + n] = v;
      if (Ch) Ch[(size_t)m * ldch + n] = (_Float16)v;
    }
  }
}

// ---------------- GATEConv edge GEMM: t = leaky(concat(xin[src], edge_attr, 0pad) @ W1^T) ----------------
__global__ __launch_bounds__(256)
void gate_gemm_kernel(const _Float16* __restrict__ xin, const _Float16* __restrict__ ea,
                      const int* __restrict__ srcIdx, const _Float16* __restrict__ W1,
                      _Float16* __restrict__ T, int E){
  __shared__ _Float16 As[16 * 224];
  __shared__ int srcs[16];
  int e0 = blockIdx.x * 16;
  if (threadIdx.x < 16){
    int e = e0 + threadIdx.x;
    srcs[threadIdx.x] = (e < E) ? srcIdx[e] : 0;
  }
  __syncthreads();
  for (int i = threadIdx.x; i < 16 * 224; i += 256){
    int r = i / 224, c = i - r * 224;
    int e = e0 + r;
    _Float16 v = (_Float16)0.f;
    if (e < E){
      if (c < 192)       v = xin[(size_t)srcs[r] * 192 + c];
      else if (c < 208)  v = ea[(size_t)e * DED + (c - 192)];
    }
    As[i] = v;
  }
  __syncthreads();
  int wave = threadIdx.x >> 5, lane = threadIdx.x & 31;
  int n0 = wave * 16;
  v8f acc = {};
  for (int k0 = 0; k0 < 224; k0 += 32){
    __builtin_prefetch(W1 + (size_t)n0 * 224 + k0 + 32, 0, 1);
    v16h a = load_a_f16(As, 224, 0, k0, 16);
    v16h b = load_b_f16(W1, 224, n0, k0);
    acc = wmma32(a, b, acc);
  }
  int half = lane >> 4, n = n0 + (lane & 15);
#pragma unroll
  for (int r = 0; r < 8; ++r){
    int e = e0 + r + 8 * half;
    if (e < E) T[(size_t)e * DH + n] = (_Float16)lrelu(acc[r]);
  }
}

// ---------------- GATEConv aggregation: h[dst] += (t @ W2^T) * alpha, fused scatter ----------------
__global__ __launch_bounds__(256)
void gate_aggr_kernel(const _Float16* __restrict__ T, const _Float16* __restrict__ W2,
                      const float* __restrict__ alpha, const int* __restrict__ dstIdx,
                      float* __restrict__ H, int E){
  int wave = threadIdx.x >> 5, lane = threadIdx.x & 31;
  int e0 = blockIdx.x * 16;
  int n0 = wave * 16;
  v8f acc = {};
  for (int k0 = 0; k0 < DH; k0 += 32){
    v16h a = load_a_f16(T, DH, e0, k0, E);
    v16h b = load_b_f16(W2, DH, n0, k0);
    acc = wmma32(a, b, acc);
  }
  int half = lane >> 4, n = n0 + (lane & 15);
#pragma unroll
  for (int r = 0; r < 8; ++r){
    int e = e0 + r + 8 * half;
    if (e < E) atomicAdd(&H[(size_t)dstIdx[e] * DH + n], acc[r] * alpha[e]);
  }
}

// ---------------- fused GRU cell: xh = relu(GRU(elu(Hin), xh)) ----------------
__global__ __launch_bounds__(256)
void gru_fused_kernel(const float* __restrict__ Hin, int ldh,
                      const _Float16* __restrict__ Xh16, int ldx,
                      const float* __restrict__ XhF, int ldxf,
                      const _Float16* __restrict__ Wi, const _Float16* __restrict__ Wh,
                      const float* __restrict__ bi, const float* __restrict__ bh,
                      _Float16* __restrict__ Out16, int ldo,
                      float* __restrict__ OutF, int ldof, int M){
  __shared__ float gbuf[16 * 772];           // [16 rows][gi(384) | gh(384)] + pad
  int wave = threadIdx.x >> 5, lane = threadIdx.x & 31;
  int row0 = blockIdx.x * 16;
  for (int part = 0; part < 2; ++part){
    const _Float16* Wm = part ? Wh : Wi;
    const float*    bb = part ? bh : bi;
    for (int tt = 0; tt < 3; ++tt){
      int n0 = (wave + 8 * tt) * 16;         // 0..368
      v8f acc = {};
      for (int k0 = 0; k0 < DH; k0 += 32){
        v16h a = part ? load_a_f16(Xh16, ldx, row0, k0, M)
                      : load_a_elu_f32(Hin, ldh, row0, k0, M);
        v16h b = load_b_f16(Wm, DH, n0, k0);
        acc = wmma32(a, b, acc);
      }
      int half = lane >> 4, n = n0 + (lane & 15);
      float bv = bb[n];
#pragma unroll
      for (int r = 0; r < 8; ++r)
        gbuf[(r + 8 * half) * 772 + part * 384 + n] = acc[r] + bv;
    }
  }
  __syncthreads();
  for (int i = threadIdx.x; i < 16 * DH; i += 256){
    int m = i >> 7, c = i & 127;
    int row = row0 + m;
    if (row < M){
      float* g = &gbuf[m * 772];
      float r_ = sigm(g[c]       + g[384 + c]);
      float z  = sigm(g[128 + c] + g[512 + c]);
      float nn = tanhf(g[256 + c] + r_ * g[640 + c]);
      float hp = XhF[(size_t)row * ldxf + c];
      float v  = (1.f - z) * nn + z * hp;
      v = v > 0.f ? v : 0.f;                 // relu wrapper
      OutF[(size_t)row * ldof + c] = v;
      Out16[(size_t)row * ldo + c] = (_Float16)v;
    }
  }
}

// ---------------- dots / attention / softmax helpers ----------------
__global__ void rowdot_kernel(const _Float16* __restrict__ A, int lda,
                              const float* __restrict__ v, float* __restrict__ out, int M, int K){
  int m = blockIdx.x * blockDim.x + threadIdx.x;
  if (m >= M) return;
  const _Float16* p = A + (size_t)m * lda;
  float acc = 0.f;
  for (int k = 0; k < K; ++k) acc += (float)p[k] * v[k];
  out[m] = acc;
}

__global__ void edge_att_gate_kernel(const _Float16* __restrict__ T, const float* __restrict__ attl,
                                     const float* __restrict__ xr, const int* __restrict__ dstIdx,
                                     float* __restrict__ a, int E){
  int e = blockIdx.x * blockDim.x + threadIdx.x;
  if (e >= E) return;
  const _Float16* p = T + (size_t)e * DH;
  float acc = 0.f;
#pragma unroll 8
  for (int k = 0; k < DH; ++k) acc += (float)p[k] * attl[k];
  a[e] = lrelu(acc + xr[dstIdx[e]]);
}

__global__ void edge_att_atom_kernel(const float* __restrict__ as, const float* __restrict__ ad,
                                     const int* __restrict__ srcIdx, const int* __restrict__ dstIdx,
                                     float* __restrict__ a, int E){
  int e = blockIdx.x * blockDim.x + threadIdx.x;
  if (e >= E) return;
  a[e] = lrelu(as[srcIdx[e]] + ad[dstIdx[e]]);
}

__global__ void node_att_mol_kernel(const float* __restrict__ as, const float* __restrict__ ad,
                                    const int* __restrict__ batch, float* __restrict__ a, int N){
  int n = blockIdx.x * blockDim.x + threadIdx.x;
  if (n >= N) return;
  a[n] = lrelu(as[n] + ad[batch[n]]);
}

__global__ void seg_max_kernel(const float* __restrict__ a, const int* __restrict__ seg,
                               float* __restrict__ m, int n){
  int i = blockIdx.x * blockDim.x + threadIdx.x;
  if (i >= n) return;
  float v = a[i];
  float* addr = &m[seg[i]];
  if (v >= 0.f) atomicMax((int*)addr, __float_as_int(v));
  else          atomicMin((unsigned int*)addr, __float_as_uint(v));
}

__global__ void seg_expsum_kernel(float* __restrict__ a, const int* __restrict__ seg,
                                  const float* __restrict__ m, float* __restrict__ s, int n){
  int i = blockIdx.x * blockDim.x + threadIdx.x;
  if (i >= n) return;
  int sg = seg[i];
  float ex = __expf(a[i] - m[sg]);
  a[i] = ex;
  atomicAdd(&s[sg], ex);
}

__global__ void seg_norm_kernel(float* __restrict__ a, const int* __restrict__ seg,
                                const float* __restrict__ s, int n){
  int i = blockIdx.x * blockDim.x + threadIdx.x;
  if (i >= n) return;
  a[i] = a[i] / (s[seg[i]] + 1e-16f);
}

// h[dst] += hs16[src] * alpha   (one edge per wave, 4 cols per lane)
__global__ void atom_scatter_kernel(const _Float16* __restrict__ hs, const float* __restrict__ alpha,
                                    const int* __restrict__ srcIdx, const int* __restrict__ dstIdx,
                                    float* __restrict__ H, int E){
  int e = blockIdx.x * 8 + (threadIdx.x >> 5);
  if (e >= E) return;
  int lane = threadIdx.x & 31;
  float al = alpha[e];
  const _Float16* p = hs + (size_t)srcIdx[e] * DH;
  float* h = H + (size_t)dstIdx[e] * DH;
  for (int c = lane; c < DH; c += 32) atomicAdd(&h[c], (float)p[c] * al);
}

// H[batch[n]] += hs16[n] * alpha[n]
__global__ void mol_scatter_kernel(const _Float16* __restrict__ hs, const float* __restrict__ alpha,
                                   const int* __restrict__ batch, float* __restrict__ H, int N){
  int n = blockIdx.x * 8 + (threadIdx.x >> 5);
  if (n >= N) return;
  int lane = threadIdx.x & 31;
  float al = alpha[n];
  const _Float16* p = hs + (size_t)n * DH;
  float* h = H + (size_t)batch[n] * DH;
  for (int c = lane; c < DH; c += 32) atomicAdd(&h[c], (float)p[c] * al);
}

// out[batch[n]] += xh[n]
__global__ void mol_pool_kernel(const float* __restrict__ xh, const int* __restrict__ batch,
                                float* __restrict__ out, int N){
  int n = blockIdx.x * 8 + (threadIdx.x >> 5);
  if (n >= N) return;
  int lane = threadIdx.x & 31;
  const float* p = xh + (size_t)n * DH;
  float* o = out + (size_t)batch[n] * DH;
  for (int c = lane; c < DH; c += 32) atomicAdd(&o[c], p[c]);
}

__global__ void relu_cvt_kernel(float* __restrict__ f, _Float16* __restrict__ h, int n){
  int i = blockIdx.x * blockDim.x + threadIdx.x;
  if (i >= n) return;
  float v = f[i]; v = v > 0.f ? v : 0.f;
  f[i] = v; h[i] = (_Float16)v;
}

__global__ void pred_kernel(const float* __restrict__ out, const float* __restrict__ w,
                            const float* __restrict__ b, float* __restrict__ y){
  int i = blockIdx.x * blockDim.x + threadIdx.x;
  if (i >= NB * DOUT) return;
  int bb = i >> 3, o = i & 7;
  const float* p = out + (size_t)bb * DH;
  const float* wp = w + (size_t)o * DH;
  float acc = b[o];
  for (int k = 0; k < DH; ++k) acc += p[k] * wp[k];
  y[i] = acc;
}

// =====================================================================================
extern "C" void kernel_launch(void* const* d_in, const int* in_sizes, int n_in,
                              void* d_out, int out_size, void* d_ws, size_t ws_size,
                              hipStream_t stream) {
  const float* x          = (const float*)d_in[0];
  const float* edge_attr  = (const float*)d_in[1];
  const int*   edge_index = (const int*)d_in[2];
  const int*   batch      = (const int*)d_in[3];
  const float* lin_node_w = (const float*)d_in[4];
  const float* lin_node_b = (const float*)d_in[5];
  const float* lin1_w     = (const float*)d_in[6];
  const float* lin1_b     = (const float*)d_in[7];
  const float* gate_att_l = (const float*)d_in[8];
  const float* gate_att_r = (const float*)d_in[9];
  const float* gate_lin1_w= (const float*)d_in[10];
  const float* gate_lin2_w= (const float*)d_in[11];
  const float* gate_bias  = (const float*)d_in[12];
  const float* gru0_wi    = (const float*)d_in[13];
  const float* gru0_wh    = (const float*)d_in[14];
  const float* gru0_bi    = (const float*)d_in[15];
  const float* gru0_bh    = (const float*)d_in[16];
  const float* atom_w     = (const float*)d_in[17];
  const float* atom_att_src=(const float*)d_in[18];
  const float* atom_att_dst=(const float*)d_in[19];
  const float* atom_bias  = (const float*)d_in[20];
  const float* atom_gru_wi= (const float*)d_in[21];
  const float* atom_gru_wh= (const float*)d_in[22];
  const float* atom_gru_bi= (const float*)d_in[23];
  const float* atom_gru_bh= (const float*)d_in[24];
  const float* mol_w      = (const float*)d_in[25];
  const float* mol_att_src= (const float*)d_in[26];
  const float* mol_att_dst= (const float*)d_in[27];
  const float* mol_bias   = (const float*)d_in[28];
  const float* mol_gru_wi = (const float*)d_in[29];
  const float* mol_gru_wh = (const float*)d_in[30];
  const float* mol_gru_bi = (const float*)d_in[31];
  const float* mol_gru_bh = (const float*)d_in[32];
  const float* pred_w     = (const float*)d_in[33];
  const float* pred_b     = (const float*)d_in[34];
  float* y = (float*)d_out;

  const int N = N_NODES, E = N_EDGES, B = NB;
  const int* src = edge_index;
  const int* dst = edge_index + E;

  // ---- workspace bump allocator ----
  char* base = (char*)d_ws; size_t off = 0;
  auto alloc = [&](size_t bytes)->void*{ void* p = base + off; off = (off + bytes + 255) & ~(size_t)255; return p; };
  _Float16* x16    = (_Float16*)alloc((size_t)N * DIN * 2);
  _Float16* ea16   = (_Float16*)alloc((size_t)E * DED * 2);
  _Float16* xin16  = (_Float16*)alloc((size_t)N * 192 * 2);
  float*    xhF    = (float*)   alloc((size_t)N * DH * 4);
  _Float16* lin1w16= (_Float16*)alloc(128 * 64 * 2);
  _Float16* w1_16  = (_Float16*)alloc(128 * 224 * 2);
  _Float16* w2_16  = (_Float16*)alloc(128 * 128 * 2);
  _Float16* atomw16= (_Float16*)alloc(2 * 128 * 192 * 2);
  _Float16* g0wi16 = (_Float16*)alloc(384 * 128 * 2);
  _Float16* g0wh16 = (_Float16*)alloc(384 * 128 * 2);
  _Float16* agwi16 = (_Float16*)alloc(2 * 384 * 128 * 2);
  _Float16* agwh16 = (_Float16*)alloc(2 * 384 * 128 * 2);
  _Float16* mgwi16 = (_Float16*)alloc(384 * 128 * 2);
  _Float16* mgwh16 = (_Float16*)alloc(384 * 128 * 2);
  _Float16* molw16 = (_Float16*)alloc(128 * 128 * 2);
  _Float16* T16    = (_Float16*)alloc((size_t)E * DH * 2);
  float*    aE     = (float*)   alloc((size_t)E * 4);
  float*    mN     = (float*)   alloc((size_t)N * 4);
  float*    sN     = (float*)   alloc((size_t)N * 4);
  float*    hF     = (float*)   alloc((size_t)N * DH * 4);
  _Float16* hs16   = (_Float16*)alloc((size_t)N * DH * 2);
  float*    asrcN  = (float*)   alloc((size_t)N * 4);
  float*    adstN  = (float*)   alloc((size_t)N * 4);
  float*    outF   = (float*)   alloc((size_t)B * DH * 4);
  _Float16* out16  = (_Float16*)alloc((size_t)B * DH * 2);
  _Float16* hd16   = (_Float16*)alloc((size_t)B * DH * 2);
  float*    hmolF  = (float*)   alloc((size_t)B * DH * 4);
  float*    adB    = (float*)   alloc((size_t)B * 4);
  float*    mB     = (float*)   alloc((size_t)B * 4);
  float*    sB     = (float*)   alloc((size_t)B * 4);

  auto cast = [&](const float* s, _Float16* d, int rows, int sc, int dc){
    int tot = rows * dc;
    cast_pad_kernel<<<(tot + 255) / 256, 256, 0, stream>>>(s, d, rows, sc, dc);
  };
  auto fill = [&](float* p, float v, int n){
    fill_kernel<<<(n + 255) / 256, 256, 0, stream>>>(p, v, n);
  };
  auto gemm = [&](const _Float16* A, int lda, const _Float16* W, int ldw, const float* bias,
                  float* Cf, int ldc, _Float16* Ch, int ldch, int M, int Nout, int K, int act){
    dim3 g((M + 15) / 16, (Nout + 127) / 128);
    gemm_f16_kernel<<<g, 256, 0, stream>>>(A, lda, W, ldw, bias, Cf, ldc, Ch, ldch, M, Nout, K, act);
  };
  auto softmax = [&](float* a, const int* seg, float* m, float* s, int n, int nseg){
    fill(m, -1e30f, nseg); fill(s, 0.f, nseg);
    int g = (n + 255) / 256;
    seg_max_kernel   <<<g, 256, 0, stream>>>(a, seg, m, n);
    seg_expsum_kernel<<<g, 256, 0, stream>>>(a, seg, m, s, n);
    seg_norm_kernel  <<<g, 256, 0, stream>>>(a, seg, s, n);
  };
  auto gru = [&](const float* Hin, const _Float16* Xh16, int ldx, float* XhFp, int ldxf,
                 const _Float16* Wi, const _Float16* Wh, const float* bi, const float* bh,
                 _Float16* O16, int ldo, float* OF, int ldof, int M){
    gru_fused_kernel<<<(M + 15) / 16, 256, 0, stream>>>(Hin, DH, Xh16, ldx, XhFp, ldxf,
                                                        Wi, Wh, bi, bh, O16, ldo, OF, ldof, M);
  };

  // ---- weight / input casts ----
  cast(x, x16, N, DIN, DIN);
  cast(edge_attr, ea16, E, DED, DED);
  cast(lin1_w, lin1w16, 128, 64, 64);
  cast(gate_lin1_w, w1_16, 128, 208, 224);   // zero-pad K to 224
  cast(gate_lin2_w, w2_16, 128, 128, 128);
  cast(atom_w, atomw16, 2 * 128, 192, 192);
  cast(gru0_wi, g0wi16, 384, 128, 128);
  cast(gru0_wh, g0wh16, 384, 128, 128);
  cast(atom_gru_wi, agwi16, 2 * 384, 128, 128);
  cast(atom_gru_wh, agwh16, 2 * 384, 128, 128);
  cast(mol_gru_wi, mgwi16, 384, 128, 128);
  cast(mol_gru_wh, mgwh16, 384, 128, 128);
  cast(mol_w, molw16, 128, 128, 128);

  // ---- node prep: xin = [x*nw | leaky(x@lin1^T+b)], xhF f32 copy ----
  node_gate_kernel<<<(N + 255) / 256, 256, 0, stream>>>(x, lin_node_w, lin_node_b, xin16, N);
  gemm(x16, DIN, lin1w16, DIN, lin1_b, xhF, DH, xin16 + 64, 192, N, DH, DIN, 1);

  // ---- GATEConv ----
  rowdot_kernel<<<(N + 255) / 256, 256, 0, stream>>>(xin16, 192, gate_att_r, asrcN, N, 192); // xr
  gate_gemm_kernel<<<(E + 15) / 16, 256, 0, stream>>>(xin16, ea16, src, w1_16, T16, E);
  edge_att_gate_kernel<<<(E + 255) / 256, 256, 0, stream>>>(T16, gate_att_l, asrcN, dst, aE, E);
  softmax(aE, dst, mN, sN, E, N);
  bias_init_kernel<<<((size_t)N * DH + 255) / 256, 256, 0, stream>>>(hF, gate_bias, N, DH);
  gate_aggr_kernel<<<(E + 15) / 16, 256, 0, stream>>>(T16, w2_16, aE, dst, hF, E);
  gru(hF, xin16 + 64, 192, xhF, DH, g0wi16, g0wh16, gru0_bi, gru0_bh, xin16 + 64, 192, xhF, DH, N);

  // ---- atom-level GATConv layers ----
  for (int l = 0; l < 2; ++l){
    gemm(xin16, 192, atomw16 + (size_t)l * 128 * 192, 192, nullptr,
         nullptr, 0, hs16, DH, N, DH, 192, 0);
    rowdot_kernel<<<(N + 255) / 256, 256, 0, stream>>>(hs16, DH, atom_att_src + l * DH, asrcN, N, DH);
    rowdot_kernel<<<(N + 255) / 256, 256, 0, stream>>>(hs16, DH, atom_att_dst + l * DH, adstN, N, DH);
    edge_att_atom_kernel<<<(E + 255) / 256, 256, 0, stream>>>(asrcN, adstN, src, dst, aE, E);
    softmax(aE, dst, mN, sN, E, N);
    bias_init_kernel<<<((size_t)N * DH + 255) / 256, 256, 0, stream>>>(hF, atom_bias + l * DH, N, DH);
    atom_scatter_kernel<<<(E + 7) / 8, 256, 0, stream>>>(hs16, aE, src, dst, hF, E);
    gru(hF, xin16 + 64, 192, xhF, DH,
        agwi16 + (size_t)l * 384 * 128, agwh16 + (size_t)l * 384 * 128,
        atom_gru_bi + l * 384, atom_gru_bh + l * 384,
        xin16 + 64, 192, xhF, DH, N);
  }

  // ---- molecule-level readout ----
  fill(outF, 0.f, B * DH);
  mol_pool_kernel<<<(N + 7) / 8, 256, 0, stream>>>(xhF, batch, outF, N);
  relu_cvt_kernel<<<(B * DH + 255) / 256, 256, 0, stream>>>(outF, out16, B * DH);
  gemm(xin16 + 64, 192, molw16, DH, nullptr, nullptr, 0, hs16, DH, N, DH, DH, 0); // hs = xh@mol_w^T
  rowdot_kernel<<<(N + 255) / 256, 256, 0, stream>>>(hs16, DH, mol_att_src, asrcN, N, DH);
  for (int t = 0; t < 2; ++t){
    gemm(out16, DH, molw16, DH, nullptr, nullptr, 0, hd16, DH, B, DH, DH, 0);     // hd = out@mol_w^T
    rowdot_kernel<<<(B + 255) / 256, 256, 0, stream>>>(hd16, DH, mol_att_dst, adB, B, DH);
    node_att_mol_kernel<<<(N + 255) / 256, 256, 0, stream>>>(asrcN, adB, batch, adstN, N);
    softmax(adstN, batch, mB, sB, N, B);
    bias_init_kernel<<<(B * DH + 255) / 256, 256, 0, stream>>>(hmolF, mol_bias, B, DH);
    mol_scatter_kernel<<<(N + 7) / 8, 256, 0, stream>>>(hs16, adstN, batch, hmolF, N);
    gru(hmolF, out16, DH, outF, DH, mgwi16, mgwh16, mol_gru_bi, mol_gru_bh,
        out16, DH, outF, DH, B);
  }
  pred_kernel<<<(B * DOUT + 255) / 256, 256, 0, stream>>>(outF, pred_w, pred_b, y);
}
